// MultiheadSelfAttention_45432164057648
// MI455X (gfx1250) — matrix operand order repaired and checked by reading, not compiled, not run on hardware
//
#include <hip/hip_runtime.h>

// ---------------------------------------------------------------------------
// Multi-head self-attention for MI455X (gfx1250, wave32, WMMA).
// B=2, S=2048, E=1024, H=16, D=64.
//   Kernel 0a: convert x f32 -> f16.
//   Kernel 0b: convert+transpose Wq/Wk/Wv f32 -> f16 W^T (LDS tile transpose).
//   Kernel 1 : QKV GEMM, f16 WMMA, double-buffered ASYNC global->LDS staging
//              (global_load_async_to_lds_b128 + s_wait_asynccnt), one barrier
//              per pipeline phase, DMA overlapped with WMMA, B-fragment
//              loads software-pipelined one iteration ahead.
//   Kernel 2 : flash attention, one wave per (b,h,16-query tile), online
//              softmax, WMMA for Q*K^T and P*V, LDS round-trip for P layout,
//              K/V fragments batch-preloaded ahead of the WMMA groups.
// ---------------------------------------------------------------------------

typedef __attribute__((ext_vector_type(4)))  _Float16 v4h;
typedef __attribute__((ext_vector_type(8)))  _Float16 v8h;
typedef __attribute__((ext_vector_type(16))) _Float16 v16h;
typedef __attribute__((ext_vector_type(8)))  float    v8f;

#define BB 2
#define SS 2048
#define EE 1024
#define HH 16
#define DD 64

#define USE_ASYNC_G2L 1

__device__ __forceinline__ v16h cat16(v8h a, v8h b) {
  return __builtin_shufflevector(a, b, 0,1,2,3,4,5,6,7,8,9,10,11,12,13,14,15);
}

// 16-byte global -> LDS copy. Async path: vdst = wave-relative LDS byte
// address (low 32 bits of the flat shared pointer per the ISA aperture rule),
// vaddr = 64-bit global address, GV mode (saddr = off).
__device__ __forceinline__ void g2l_b128(const _Float16* g, _Float16* l) {
#if USE_ASYNC_G2L
  const unsigned loff = (unsigned)(unsigned long long)l;
  asm volatile("global_load_async_to_lds_b128 %0, %1, off"
               :: "v"(loff), "v"(g) : "memory");
#else
  *(v8h*)l = *(const v8h*)g;
#endif
}
__device__ __forceinline__ void async_wait0() {
#if USE_ASYNC_G2L
  asm volatile("s_wait_asynccnt 0" ::: "memory");
#endif
}

// ---------------------------------------------------------------------------
// Kernel 0a: x f32 -> f16, elementwise.
// ---------------------------------------------------------------------------
__global__ __launch_bounds__(256) void cvt_x_kernel(
    const float* __restrict__ x, _Float16* __restrict__ xh)
{
  const size_t i = ((size_t)blockIdx.x * 256 + threadIdx.x) * 4;
  const float4 f = *(const float4*)(x + i);
  v4h hv = { (_Float16)f.x, (_Float16)f.y, (_Float16)f.z, (_Float16)f.w };
  *(v4h*)(xh + i) = hv;
}

// ---------------------------------------------------------------------------
// Kernel 0b: W f32 [k][n] -> WT f16 [z][n][k], 32x32 LDS tile transpose.
// grid = (E/32, E/32, 3), block = 256.
// ---------------------------------------------------------------------------
__global__ __launch_bounds__(256) void cvt_w_kernel(
    const float* __restrict__ Wq, const float* __restrict__ Wk,
    const float* __restrict__ Wv, _Float16* __restrict__ WT)
{
  __shared__ _Float16 t[32][36];
  const int z = blockIdx.z;
  const float* __restrict__ W = (z == 0) ? Wq : ((z == 1) ? Wk : Wv);
  const int kBase = blockIdx.x * 32, nBase = blockIdx.y * 32;
  const int r  = threadIdx.x >> 3;
  const int c4 = (threadIdx.x & 7) * 4;

  const float4 f = *(const float4*)(W + (size_t)(kBase + r) * EE + nBase + c4);
  t[c4 + 0][r] = (_Float16)f.x;
  t[c4 + 1][r] = (_Float16)f.y;
  t[c4 + 2][r] = (_Float16)f.z;
  t[c4 + 3][r] = (_Float16)f.w;
  __syncthreads();

  v4h o = { t[r][c4 + 0], t[r][c4 + 1], t[r][c4 + 2], t[r][c4 + 3] };
  *(v4h*)(WT + (size_t)z * EE * EE + (size_t)(nBase + r) * EE + kBase + c4) = o;
}

// ---------------------------------------------------------------------------
// Kernel 1: QKV GEMM. grid = (M/128, N/128, 3), block = 256 (8 waves).
// Wave w computes rows [16w,16w+16) x the full 128-col N tile (8 WMMA/phase).
// Double-buffered: async DMA of tile k+1 overlaps WMMA of tile k.
// ---------------------------------------------------------------------------
__global__ __launch_bounds__(256) void qkv_gemm_kernel(
    const _Float16* __restrict__ xh, const _Float16* __restrict__ WT,
    _Float16* __restrict__ Qb, _Float16* __restrict__ Kb,
    _Float16* __restrict__ VTb)
{
  __shared__ _Float16 xs [2][128 * 32];   // x  tile [m][k]
  __shared__ _Float16 ws2[2][128 * 32];   // W^T tile [n][k]

  const int tid   = threadIdx.x;
  const int lane  = tid & 31;
  const int wv    = tid >> 5;
  const int lmod  = lane & 15;
  const int lhalf = lane >> 4;
  const int mBase = blockIdx.x * 128;
  const int nBase = blockIdx.y * 128;
  const int z     = blockIdx.z;
  const _Float16* __restrict__ Wz = WT + (size_t)z * EE * EE;

  const v8f zero8 = {0.f,0.f,0.f,0.f,0.f,0.f,0.f,0.f};
  v8f acc[8];
#pragma unroll
  for (int j = 0; j < 8; ++j) acc[j] = zero8;

  // Stage one 128x32 f16 tile pair: 512 b128 DMAs per tile, 2+2 per thread.
  auto issueCopies = [&](int k0, int buf) {
#pragma unroll
    for (int i = 0; i < 2; ++i) {
      const int lin = tid + 256 * i;     // 0..511
      const int row = lin >> 2;          // 0..127
      const int seg = (lin & 3) * 8;     // 0,8,16,24
      g2l_b128(xh + (size_t)(mBase + row) * EE + k0 + seg, &xs [buf][row * 32 + seg]);
      g2l_b128(Wz + (size_t)(nBase + row) * EE + k0 + seg, &ws2[buf][row * 32 + seg]);
    }
  };

  // 8 WMMAs from one staged tile pair; B fragment j+1 preloaded before WMMA j.
  auto computeTile = [&](int buf) {
    // A fragment (16x32): M = lane%16; lane<16 -> K 0-15,
    // lane>=16 -> K 16-31.
    const _Float16* ar = &xs[buf][(16 * wv + lmod) * 32];
    const v8h a0 = *(const v8h*)(ar + 16 * lhalf);
    const v8h a1 = *(const v8h*)(ar + 16 * lhalf + 8);
    const v16h A = cat16(a0, a1);

    // B fragment (32x16): N = lane%16; lanes 0-15 K 0-15, lanes 16-31 K 16-31.
    const _Float16* br0 = &ws2[buf][lmod * 32 + 16 * lhalf];
    v8h b0 = *(const v8h*)(br0);
    v8h b1 = *(const v8h*)(br0 + 8);
#pragma unroll
    for (int j = 0; j < 8; ++j) {
      const v16h Bf = cat16(b0, b1);
      if (j < 7) {
        const _Float16* br = &ws2[buf][(16 * (j + 1) + lmod) * 32 + 16 * lhalf];
        b0 = *(const v8h*)(br);
        b1 = *(const v8h*)(br + 8);
      }
      acc[j] = __builtin_amdgcn_wmma_f32_16x16x32_f16(
          false, A, false, Bf, (short)0, acc[j], false, false);
    }
  };

  // Software pipeline: wait(own DMA) -> barrier(publish) -> issue next -> compute.
  issueCopies(0, 0);
  for (int k0 = 0; k0 < EE; k0 += 64) {
    async_wait0();
    __syncthreads();
    if (k0 + 32 < EE) issueCopies(k0 + 32, 1);
    computeTile(0);

    async_wait0();
    __syncthreads();
    if (k0 + 64 < EE) issueCopies(k0 + 64, 0);
    computeTile(1);
  }

  // Store. C layout: M = r + 8*(lane/16), N = lane%16. Uniform z branch only.
  const int b  = mBase >> 11;                            // M / S (tile-constant)
  const int sB = (mBase & (SS - 1)) + 16 * wv + 8 * lhalf;
  if (z == 2) {
    // V^T [b,h,d,s]: 8 rows are contiguous in s -> one packed v8h store each.
#pragma unroll
    for (int j = 0; j < 8; ++j) {
      const int h = 2 * blockIdx.y + (j >> 2);
      const int d = 16 * (j & 3) + lmod;
      v8h hv;
#pragma unroll
      for (int r = 0; r < 8; ++r) hv[r] = (_Float16)acc[j][r];
      *(v8h*)(VTb + ((size_t)(b * HH + h) * DD + d) * SS + sB) = hv;
    }
  } else {
    _Float16* __restrict__ QK = (z == 0) ? Qb : Kb;
#pragma unroll
    for (int j = 0; j < 8; ++j) {
      const int h = 2 * blockIdx.y + (j >> 2);
      const int d = 16 * (j & 3) + lmod;
      _Float16* dst = QK + ((size_t)(b * HH + h) * SS + sB) * DD + d;
#pragma unroll
      for (int r = 0; r < 8; ++r) dst[(size_t)r * DD] = (_Float16)acc[j][r];
    }
  }
}

// ---------------------------------------------------------------------------
// Kernel 2: flash attention. One wave per (b,h,qtile). Block = 8 waves.
// ---------------------------------------------------------------------------
__global__ __launch_bounds__(256) void attn_kernel(
    const _Float16* __restrict__ Qb, const _Float16* __restrict__ Kb,
    const _Float16* __restrict__ VTb, float* __restrict__ out)
{
  __shared__ _Float16 pl[8 * 16 * 32];   // per-wave 16x32 f16 P tile

  const int tid   = threadIdx.x;
  const int lane  = tid & 31;
  const int wv    = tid >> 5;
  const int lmod  = lane & 15;
  const int lhalf = lane >> 4;

  const int wid   = blockIdx.x * 8 + wv;   // 0..4095
  const int bh    = wid >> 7;              // b*H + h
  const int qbase = (wid & 127) * 16;

  const _Float16* __restrict__ Qh = Qb  + (size_t)bh * SS * DD;
  const _Float16* __restrict__ Kh = Kb  + (size_t)bh * SS * DD;
  const _Float16* __restrict__ Vh = VTb + (size_t)bh * DD * SS;
  _Float16* P = &pl[wv * 16 * 32];

  // Q A-fragments for d chunks 0-31 and 32-63 (loop invariant).
  const _Float16* qr = Qh + (size_t)(qbase + lmod) * DD;
  v16h AQ[2];
#pragma unroll
  for (int c = 0; c < 2; ++c) {
    const v8h p0 = *(const v8h*)(qr + 32 * c + 16 * lhalf);
    const v8h p1 = *(const v8h*)(qr + 32 * c + 16 * lhalf + 8);
    AQ[c] = cat16(p0, p1);
  }

  const v8f zero8 = {0.f,0.f,0.f,0.f,0.f,0.f,0.f,0.f};
  v8f Oa[4];
#pragma unroll
  for (int j = 0; j < 4; ++j) Oa[j] = zero8;
  v8f mrow, lrow;
#pragma unroll
  for (int r = 0; r < 8; ++r) { mrow[r] = -3.0e38f; lrow[r] = 0.f; }

  for (int kc = 0; kc < SS / 32; ++kc) {
    const int kb = kc * 32;

    // Prefetch next K/V tiles (global_prefetch_b8).
    if (kc + 1 < SS / 32) {
      const int kn = kb + 32;
      __builtin_prefetch(Kh + (size_t)(kn + lmod) * DD + 16 * lhalf, 0, 1);
      __builtin_prefetch(Kh + (size_t)(kn + 16 + lmod) * DD + 16 * lhalf, 0, 1);
#pragma unroll
      for (int j = 0; j < 4; ++j)
        __builtin_prefetch(Vh + (size_t)(16 * j + lmod) * SS + kn + 16 * lhalf, 0, 1);
    }

    // --- scores: batch-load all 4 K B-fragments, then 4 WMMAs ---
    v16h BK[2][2];   // [key subtile j2][d chunk c]
#pragma unroll
    for (int j2 = 0; j2 < 2; ++j2)
#pragma unroll
      for (int c = 0; c < 2; ++c) {
        const _Float16* kr =
            Kh + (size_t)(kb + 16 * j2 + lmod) * DD + 32 * c + 16 * lhalf;
        BK[j2][c] = cat16(*(const v8h*)(kr), *(const v8h*)(kr + 8));
      }
    v8f sc[2];
#pragma unroll
    for (int j2 = 0; j2 < 2; ++j2) {
      v8f s = zero8;
#pragma unroll
      for (int c = 0; c < 2; ++c)
        s = __builtin_amdgcn_wmma_f32_16x16x32_f16(
            false, AQ[c], false, BK[j2][c], (short)0, s, false, false);
#pragma unroll
      for (int r = 0; r < 8; ++r) s[r] *= 0.125f;   // 1/sqrt(64)
      sc[j2] = s;
    }

    // --- online softmax. Row max across 16 lanes of each half. ---
    v8f tmax;
#pragma unroll
    for (int r = 0; r < 8; ++r) tmax[r] = fmaxf(sc[0][r], sc[1][r]);
#pragma unroll
    for (int r = 0; r < 8; ++r) {
      float t = tmax[r];
      t = fmaxf(t, __shfl_xor(t, 1));
      t = fmaxf(t, __shfl_xor(t, 2));
      t = fmaxf(t, __shfl_xor(t, 4));
      t = fmaxf(t, __shfl_xor(t, 8));
      tmax[r] = t;
    }
    v8f mnew, corr;
#pragma unroll
    for (int r = 0; r < 8; ++r) {
      mnew[r] = fmaxf(mrow[r], tmax[r]);
      corr[r] = __expf(mrow[r] - mnew[r]);
      mrow[r] = mnew[r];
    }
    v8f p0, p1;
#pragma unroll
    for (int r = 0; r < 8; ++r) {
      p0[r] = __expf(sc[0][r] - mnew[r]);
      p1[r] = __expf(sc[1][r] - mnew[r]);
    }
    v8f rs;
#pragma unroll
    for (int r = 0; r < 8; ++r) {
      float t = p0[r] + p1[r];
      t += __shfl_xor(t, 1);
      t += __shfl_xor(t, 2);
      t += __shfl_xor(t, 4);
      t += __shfl_xor(t, 8);
      rs[r] = t;
    }
#pragma unroll
    for (int r = 0; r < 8; ++r) lrow[r] = lrow[r] * corr[r] + rs[r];
#pragma unroll
    for (int j = 0; j < 4; ++j)
#pragma unroll
      for (int r = 0; r < 8; ++r) Oa[j][r] *= corr[r];

    // --- P: C-layout -> LDS row-major [16 q][32 k] -> A fragment ---
#pragma unroll
    for (int r = 0; r < 8; ++r) {
      const int row = r + 8 * lhalf;
      P[row * 32 + lmod]      = (_Float16)p0[r];
      P[row * 32 + 16 + lmod] = (_Float16)p1[r];
    }
    const _Float16* pr = &P[lmod * 32];
    const v8h q0 = *(const v8h*)(pr + 16 * lhalf);
    const v8h q1 = *(const v8h*)(pr + 16 * lhalf + 8);
    const v16h AP = cat16(q0, q1);

    // --- O += P @ V: batch-load all 4 V B-fragments, then 4 WMMAs ---
    v16h BV[4];
#pragma unroll
    for (int j = 0; j < 4; ++j) {
      const _Float16* vr =
          Vh + (size_t)(16 * j + lmod) * SS + kb + 16 * lhalf;
      BV[j] = cat16(*(const v8h*)(vr), *(const v8h*)(vr + 8));
    }
#pragma unroll
    for (int j = 0; j < 4; ++j)
      Oa[j] = __builtin_amdgcn_wmma_f32_16x16x32_f16(
          false, AP, false, BV[j], (short)0, Oa[j], false, false);
  }

  // Epilogue: normalize and store [b][s][h*64+d] as f32 (non-temporal:
  // output is written once and never re-read; keep Q/K/V resident in L2).
  const int h = bh & (HH - 1);
  const int b = bh >> 4;
  v8f rinv;
#pragma unroll
  for (int r = 0; r < 8; ++r) rinv[r] = 1.0f / lrow[r];
#pragma unroll
  for (int j = 0; j < 4; ++j) {
#pragma unroll
    for (int r = 0; r < 8; ++r) {
      const int s   = qbase + r + 8 * lhalf;
      const int col = h * DD + 16 * j + lmod;
      __builtin_nontemporal_store(Oa[j][r] * rinv[r],
                                  out + ((size_t)(b * SS + s)) * EE + col);
    }
  }
}

// ---------------------------------------------------------------------------
extern "C" void kernel_launch(void* const* d_in, const int* in_sizes, int n_in,
                              void* d_out, int out_size, void* d_ws, size_t ws_size,
                              hipStream_t stream) {
  const float* x  = (const float*)d_in[0];
  const float* Wq = (const float*)d_in[1];
  const float* Wk = (const float*)d_in[2];
  const float* Wv = (const float*)d_in[3];
  float* out = (float*)d_out;

  const size_t perBuf = (size_t)BB * HH * SS * DD;  // 4M f16 elems = 8 MB
  _Float16* Qb  = (_Float16*)d_ws;
  _Float16* Kb  = Qb + perBuf;
  _Float16* VTb = Kb + perBuf;
  _Float16* xh  = VTb + perBuf;                     // [B*S][E]     (4M)
  _Float16* WT  = xh + (size_t)BB * SS * EE;        // [3][E][E]    (3M)

  cvt_x_kernel<<<(BB * SS * EE / 4) / 256, 256, 0, stream>>>(x, xh);
  cvt_w_kernel<<<dim3(EE / 32, EE / 32, 3), 256, 0, stream>>>(Wq, Wk, Wv, WT);

  dim3 g1((BB * SS) / 128, EE / 128, 3);
  qkv_gemm_kernel<<<g1, 256, 0, stream>>>(xh, WT, Qb, Kb, VTb);

  const int nWaves = BB * HH * (SS / 16);           // 4096
  attn_kernel<<<nWaves / 8, 256, 0, stream>>>(Qb, Kb, VTb, out);
}
